// VectorFieldAttention_73856257622518
// MI455X (gfx1250) — matrix operand order repaired and checked by reading, not compiled, not run on hardware
//
#include <hip/hip_runtime.h>
#include <hip/hip_bf16.h>

// ---------------------------------------------------------------------------
// CDNA5 (gfx1250) WMMA 16x16x32 f16->f32, wave32 fragment layouts (ISA 7.12.2):
//   A 16x32: lane l -> m=l&15, half=l>>4; element s: k=half*16+s (this code's order, not ISA table)
//            => per lane 16 contiguous f16 at k = half*16 (matches B)
//   B 32x16: lane l -> n=l&15, half=l>>4; element s: k=half*16+s at column n (this code's order)
//            => per lane 16 contiguous f16 if B is stored N-major ([n][k])
//   C  f32 : acc[r] = C[m = r + 8*half][n = l&15]
// All fragment loads below are 16B vector loads (ds_load_b128/global_load_b128).
// ---------------------------------------------------------------------------
typedef __attribute__((ext_vector_type(16))) _Float16 v16h;
typedef __attribute__((ext_vector_type(8)))  _Float16 h8;
typedef __attribute__((ext_vector_type(8)))  float    v8f;

__device__ __forceinline__ v8f wmma_f16(v16h a, v16h b, v8f c) {
  return __builtin_amdgcn_wmma_f32_16x16x32_f16(
      /*neg_a=*/false, a, /*neg_b=*/false, b,
      /*c_mod=*/(short)0, c, /*reuse_a=*/false, /*reuse_b=*/false);
}

// A[m][k] = base[(row0+m)*ld + kbase + k], row-major MxK f16.
// Requires: ld % 8 == 0, kbase % 8 == 0, base 16B aligned.
__device__ __forceinline__ v16h load_a16(const _Float16* base, int row0, int ld, int kbase) {
  int l = threadIdx.x & 31; int m = l & 15; int half = l >> 4;
  const _Float16* p = base + (size_t)(row0 + m) * ld + kbase + half * 16;
  union { v16h v; h8 h[2]; } u;
  u.h[0] = *(const h8*)p;          // k = kbase + half*16 + 0..7
  u.h[1] = *(const h8*)(p + 8);    // k = kbase + half*16 + 8..15
  return u.v;
}

// B[k][n] = base[(col0+n)*ld + kbase + k]  (source stored N-major: [n][k] f16)
// Requires: ld % 8 == 0, kbase % 8 == 0, base 16B aligned.
__device__ __forceinline__ v16h load_b16t(const _Float16* base, int col0, int ld, int kbase) {
  int l = threadIdx.x & 31; int n = l & 15; int half = l >> 4;
  const _Float16* p = base + (size_t)(col0 + n) * ld + kbase + half * 16;
  union { v16h v; h8 h[2]; } u;
  u.h[0] = *(const h8*)p;          // k = kbase + half*16 + 0..7
  u.h[1] = *(const h8*)(p + 8);    // k = kbase + half*16 + 8..15
  return u.v;
}

// ---------------------------------------------------------------------------
// Pass 0: prepack weights to f16 in WMMA-friendly layouts (once per launch).
//   w1p  : [2][9][32][64]  (path, tap, oc, ic)  A-fragment row-major
//   w2p  : [2][9][64][32]
//   wqkvp: [384][128] f16 row-major  (B via load_b16t)
//   wop  : [128][128] f16 row-major
// ---------------------------------------------------------------------------
__global__ __launch_bounds__(256) void prepack_kernel(
    const float* __restrict__ mw1, const float* __restrict__ aw1,
    const float* __restrict__ mw2, const float* __restrict__ aw2,
    const float* __restrict__ wqkv, const float* __restrict__ wo,
    _Float16* __restrict__ w1p, _Float16* __restrict__ w2p,
    _Float16* __restrict__ wqkvp, _Float16* __restrict__ wop) {
  int i = blockIdx.x * 256 + threadIdx.x;
  if (i < 36864) {                                   // w1p
    int path = i / 18432, r = i % 18432;
    int tap = r / 2048, r2 = r % 2048;
    int oc = r2 / 64, ic = r2 % 64;
    const float* src = path ? aw1 : mw1;
    w1p[i] = (_Float16)src[(size_t)(oc * 64 + ic) * 9 + tap];
  } else if (i < 73728) {                            // w2p
    int j = i - 36864;
    int path = j / 18432, r = j % 18432;
    int tap = r / 2048, r2 = r % 2048;
    int oc = r2 / 32, ic = r2 % 32;
    const float* src = path ? aw2 : mw2;
    w2p[j] = (_Float16)src[(size_t)(oc * 32 + ic) * 9 + tap];
  } else if (i < 122880) {                           // wqkvp
    int j = i - 73728;
    wqkvp[j] = (_Float16)wqkv[j];
  } else if (i < 139264) {                           // wop
    int j = i - 122880;
    wop[j] = (_Float16)wo[j];
  }
}

// ---------------------------------------------------------------------------
// Pass 1: conv3x3 (64 -> 32) + ReLU, both paths. Implicit GEMM.
// Input tile staged N-major in LDS: sInT[row][col][ic] so B frags are b128.
// ---------------------------------------------------------------------------
__global__ __launch_bounds__(256) void conv1_kernel(
    const float* __restrict__ vf, const _Float16* __restrict__ w1p,
    const float* __restrict__ mb1, const float* __restrict__ ab1,
    _Float16* __restrict__ t1) {
  const int H = 128, W = 128, ICP = 72;   // ic padded 64->72 (16B-aligned rows)
  int tile = blockIdx.x;                   // 32 (rows/4) x 8 (cols/16)
  int path = blockIdx.y, b = blockIdx.z;
  int tx0 = (tile & 7) * 16;
  int ty0 = (tile >> 3) * 4;
  const float* b1 = path ? ab1 : mb1;
  const float* in = vf + (size_t)(b * 128 + path * 64) * H * W;

  __shared__ __align__(16) _Float16 sInT[6 * 20 * ICP];  // [row][col][ic]
  for (int i = threadIdx.x; i < 64 * 6 * 18; i += 256) {
    int ic = i / 108, rem = i % 108, r = rem / 18, c = rem % 18;
    int y = ty0 - 1 + r, x = tx0 - 1 + c;
    float v = 0.f;
    if ((unsigned)y < (unsigned)H && (unsigned)x < (unsigned)W)
      v = in[(size_t)ic * H * W + y * W + x];
    sInT[(r * 20 + c) * ICP + ic] = (_Float16)v;
  }
  __syncthreads();

  int wv = threadIdx.x >> 5;
  int ocb = (wv & 1) * 16;                 // 2 oc tiles
  int r = wv >> 1;                         // 4 pixel rows
  v8f acc = {};
#pragma unroll
  for (int tap = 0; tap < 9; ++tap) {
    int dy = tap / 3, dx = tap % 3;
    const _Float16* aw = w1p + (size_t)(path * 9 + tap) * 32 * 64;
    const _Float16* bb = sInT + ((r + dy) * 20 + dx) * ICP;
#pragma unroll
    for (int ks = 0; ks < 2; ++ks) {       // K = 64 ic
      v16h a = load_a16(aw, ocb, 64, ks * 32);
      v16h bv = load_b16t(bb, 0, ICP, ks * 32);
      acc = wmma_f16(a, bv, acc);
    }
  }
  int l = threadIdx.x & 31, n = l & 15, half = l >> 4;
  int y = ty0 + r;
  _Float16* outp = t1 + (size_t)((b * 2 + path) * 32) * H * W;
#pragma unroll
  for (int rr = 0; rr < 8; ++rr) {
    int oc = ocb + rr + half * 8;
    float v = acc[rr] + b1[oc];
    v = v > 0.f ? v : 0.f;
    outp[(size_t)oc * H * W + y * W + tx0 + n] = (_Float16)v;
  }
}

// ---------------------------------------------------------------------------
// Pass 2: conv3x3 (32 -> 64) -> combined0 (B,128,H,W) f32
// ---------------------------------------------------------------------------
__global__ __launch_bounds__(256) void conv2_kernel(
    const _Float16* __restrict__ t1, const _Float16* __restrict__ w2p,
    const float* __restrict__ mb2, const float* __restrict__ ab2,
    float* __restrict__ comb) {
  const int H = 128, W = 128, ICP = 48;   // ic padded 32->48
  int tile = blockIdx.x;
  int path = blockIdx.y, b = blockIdx.z;
  int tx0 = (tile & 7) * 16;
  int ty0 = (tile >> 3) * 4;
  const float* b2 = path ? ab2 : mb2;
  const _Float16* in = t1 + (size_t)((b * 2 + path) * 32) * H * W;

  __shared__ __align__(16) _Float16 sInT[6 * 20 * ICP];
  for (int i = threadIdx.x; i < 32 * 6 * 18; i += 256) {
    int ic = i / 108, rem = i % 108, r = rem / 18, c = rem % 18;
    int y = ty0 - 1 + r, x = tx0 - 1 + c;
    _Float16 v = (_Float16)0.f;
    if ((unsigned)y < (unsigned)H && (unsigned)x < (unsigned)W)
      v = in[(size_t)ic * H * W + y * W + x];
    sInT[(r * 20 + c) * ICP + ic] = v;
  }
  __syncthreads();

  int wv = threadIdx.x >> 5;
  int l = threadIdx.x & 31, n = l & 15, half = l >> 4;
#pragma unroll
  for (int tt = 0; tt < 2; ++tt) {
    int tid = wv * 2 + tt;                 // 16 tiles: 4 oc x 4 rows
    int ocb = (tid & 3) * 16;
    int r = tid >> 2;
    v8f acc = {};
#pragma unroll
    for (int tap = 0; tap < 9; ++tap) {
      int dy = tap / 3, dx = tap % 3;
      v16h a = load_a16(w2p + (size_t)(path * 9 + tap) * 64 * 32, ocb, 32, 0);
      v16h bv = load_b16t(sInT + ((r + dy) * 20 + dx) * ICP, 0, ICP, 0);
      acc = wmma_f16(a, bv, acc);
    }
    int y = ty0 + r;
#pragma unroll
    for (int rr = 0; rr < 8; ++rr) {
      int oc = ocb + rr + half * 8;
      comb[((size_t)(b * 128 + path * 64 + oc) * H + y) * W + tx0 + n] = acc[rr] + b2[oc];
    }
  }
}

// ---------------------------------------------------------------------------
// Pass 3: windowed MHA, one block per (window, batch). L=64, E=128, 4 heads.
// LDS (88KB): sQKV(q,k) 32K | sVt 16K | sX/sO 16K | sS 16K | sP 8K
// att layout: (win, B, E, 8, 8) f16
// ---------------------------------------------------------------------------
__global__ __launch_bounds__(256) void attn_kernel(
    const float* __restrict__ comb, const _Float16* __restrict__ wqkvp,
    const float* __restrict__ bqkv, const _Float16* __restrict__ wop,
    const float* __restrict__ bo, _Float16* __restrict__ att) {
  __shared__ __align__(16) char smem[90112];
  _Float16* sQKV = (_Float16*)smem;                           // 64 x 256 (q|k)
  _Float16* sVt  = (_Float16*)(smem + 32768);                 // 128 x 64 (d-major V)
  _Float16* sX   = (_Float16*)(smem + 32768 + 16384);         // 64 x 128 (= sO)
  _Float16* sO   = sX;
  float*    sS   = (float*)(smem + 32768 + 16384 + 16384);    // 64 x 64 f32
  _Float16* sP   = (_Float16*)(smem + 32768 + 16384 + 16384 + 16384); // 64 x 64

  int bid = blockIdx.x;
  int win = bid >> 2, b = bid & 3;
  int hi = win / 31, wi = win % 31;
  int h0 = hi * 4, w0 = wi * 4;

  // Prefetch shared projection weights (global_prefetch_b8 path)
  if (threadIdx.x < 48) __builtin_prefetch(wqkvp + threadIdx.x * 1024, 0, 3);
  else if (threadIdx.x < 64) __builtin_prefetch(wop + (threadIdx.x - 48) * 1024, 0, 3);

  // Load X tokens: token t = ty*8+tx
  for (int i = threadIdx.x; i < 64 * 128; i += 256) {
    int t = i >> 7, e = i & 127;
    int ty = t >> 3, tx = t & 7;
    sX[i] = (_Float16)comb[(((size_t)b * 128 + e) * 128 + h0 + ty) * 128 + (w0 + tx)];
  }
  __syncthreads();

  int wv = threadIdx.x >> 5;
  int l = threadIdx.x & 31, n = l & 15, half = l >> 4;
  const float qscale = 0.17677669529663687f;  // 1/sqrt(32)

  // ---- in_proj: (64x128)@(128x384); 96 tiles, 12/wave, K=4 steps ----
#pragma unroll 1
  for (int ti = 0; ti < 12; ++ti) {
    int tid = wv * 12 + ti;
    int mt = tid / 24, nt = tid % 24;
    v8f acc = {};
#pragma unroll
    for (int ks = 0; ks < 4; ++ks) {
      v16h a = load_a16(sX, mt * 16, 128, ks * 32);
      v16h bb = load_b16t(wqkvp, nt * 16, 128, ks * 32);
      acc = wmma_f16(a, bb, acc);
    }
    int j = nt * 16 + n;
    float bias = bqkv[j];
    float sc = (j < 128) ? qscale : 1.0f;     // q scaled (incl. bias) by hd^-0.5
#pragma unroll
    for (int rr = 0; rr < 8; ++rr) {
      int m = mt * 16 + rr + half * 8;
      float v = (acc[rr] + bias) * sc;
      if (j < 256) sQKV[m * 256 + j] = (_Float16)v;          // q,k token-major
      else         sVt[(j - 256) * 64 + m] = (_Float16)v;     // v d-major
    }
  }
  __syncthreads();

  // ---- per-head attention ----
#pragma unroll 1
  for (int h = 0; h < 4; ++h) {
    // S = q_h (64x32) @ k_h^T (32x64): 16 tiles, 2/wave
#pragma unroll
    for (int tt = 0; tt < 2; ++tt) {
      int tid = wv * 2 + tt;
      int mt = tid >> 2, nt = tid & 3;
      v16h a = load_a16(sQKV, mt * 16, 256, h * 32);
      v16h bb = load_b16t(sQKV, nt * 16, 256, 128 + h * 32);
      v8f acc = {};
      acc = wmma_f16(a, bb, acc);
#pragma unroll
      for (int rr = 0; rr < 8; ++rr)
        sS[(mt * 16 + rr + half * 8) * 64 + nt * 16 + n] = acc[rr];
    }
    __syncthreads();

    // softmax rows (64 lanes, one row each)
    if (threadIdx.x < 64) {
      float* row = sS + threadIdx.x * 64;
      float mx = row[0];
      for (int jj = 1; jj < 64; ++jj) mx = fmaxf(mx, row[jj]);
      float sum = 0.f;
      for (int jj = 0; jj < 64; ++jj) sum += __expf(row[jj] - mx);
      float inv = 1.f / sum;
      for (int jj = 0; jj < 64; ++jj)
        sP[threadIdx.x * 64 + jj] = (_Float16)(__expf(row[jj] - mx) * inv);
    }
    __syncthreads();

    // O_h = P (64x64) @ V_h (64x32): 8 tiles, 1/wave, K=2 steps
    {
      int mt = wv >> 1, nt = wv & 1;
      v8f acc = {};
#pragma unroll
      for (int ks = 0; ks < 2; ++ks) {
        v16h a = load_a16(sP, mt * 16, 64, ks * 32);
        v16h bb = load_b16t(sVt, h * 32 + nt * 16, 64, ks * 32);  // B[k][n]=V[t'=k][d=n]
        acc = wmma_f16(a, bb, acc);
      }
#pragma unroll
      for (int rr = 0; rr < 8; ++rr)
        sO[(mt * 16 + rr + half * 8) * 128 + h * 32 + nt * 16 + n] = (_Float16)acc[rr];
    }
    __syncthreads();
  }

  // ---- out_proj: (64x128)@(128x128) + bo -> att ----
#pragma unroll 1
  for (int ti = 0; ti < 4; ++ti) {
    int tid = wv * 4 + ti;                   // 32 tiles
    int mt = tid >> 3, nt = tid & 7;
    v8f acc = {};
#pragma unroll
    for (int ks = 0; ks < 4; ++ks) {
      v16h a = load_a16(sO, mt * 16, 128, ks * 32);
      v16h bb = load_b16t(wop, nt * 16, 128, ks * 32);
      acc = wmma_f16(a, bb, acc);
    }
    int j = nt * 16 + n;
    float bias = bo[j];
#pragma unroll
    for (int rr = 0; rr < 8; ++rr) {
      int m = mt * 16 + rr + half * 8;       // token index
      int ty = m >> 3, tx = m & 7;
      att[((((size_t)win * 4 + b) * 128 + j) * 8 + ty) * 8 + tx] = (_Float16)(acc[rr] + bias);
    }
  }
}

// ---------------------------------------------------------------------------
// Pass 4: exact parallelization of the sequential scan blend (<=4 covering
// windows per pixel, applied in global h-major order). In-place on comb.
// ---------------------------------------------------------------------------
__global__ __launch_bounds__(256) void blend_kernel(
    float* __restrict__ comb, const _Float16* __restrict__ att) {
  size_t idx = (size_t)blockIdx.x * 256 + threadIdx.x;
  if (idx >= (size_t)4 * 128 * 128 * 128) return;
  int x = idx & 127, y = (idx >> 7) & 127, e = (idx >> 14) & 127, b = (int)(idx >> 21);
  float v = comb[idx];
  int qy = y >> 2, qx = x >> 2;
  int hc[2]; int nh = 0;
  { int s = (qy - 1) * 4; if (s >= 0) hc[nh++] = s;
    s = qy * 4;           if (s <= 120) hc[nh++] = s; }
  int wc[2]; int nw = 0;
  { int s = (qx - 1) * 4; if (s >= 0) wc[nw++] = s;
    s = qx * 4;           if (s <= 120) wc[nw++] = s; }
  for (int i = 0; i < nh; ++i)
    for (int j = 0; j < nw; ++j) {
      int Hh = hc[i] >> 2, Ww = wc[j] >> 2;
      int win = Hh * 31 + Ww;
      int ty = y - hc[i], tx = x - wc[j];
      float a = (float)att[((((size_t)win * 4 + b) * 128 + e) * 8 + ty) * 8 + tx];
      v = v * 0.7f + a * 0.3f;
    }
  comb[idx] = v;
}

// ---------------------------------------------------------------------------
// Pass 5: flow_weights = sigmoid(conv1x1(enhanced)); out = gated + residual
// ---------------------------------------------------------------------------
__global__ __launch_bounds__(256) void final_kernel(
    const float* __restrict__ comb, const float* __restrict__ vf,
    const float* __restrict__ fcw, const float* __restrict__ fcb,
    float* __restrict__ out) {
  __shared__ float sE[128 * 64];             // 32 KB: 128 ch x 64 pixels
  int b = blockIdx.y;
  int tile = blockIdx.x;                     // 128 rows x 2 half-rows
  int y = tile >> 1, x0 = (tile & 1) * 64;
  for (int i = threadIdx.x; i < 128 * 64; i += 256) {
    int e = i >> 6, p = i & 63;
    sE[i] = comb[(((size_t)b * 128 + e) * 128 + y) * 128 + x0 + p];
  }
  __syncthreads();
  int p = threadIdx.x & 63, cg = threadIdx.x >> 6;
  for (int c = cg * 16; c < cg * 16 + 16; ++c) {
    float s = fcb[c];
#pragma unroll 4
    for (int e = 0; e < 128; ++e) s += fcw[c * 128 + e] * sE[e * 64 + p];
    float fw = 1.f / (1.f + __expf(-s));
    size_t o1 = (((size_t)b * 128 + c) * 128 + y) * 128 + x0 + p;
    size_t o2 = (((size_t)b * 128 + 64 + c) * 128 + y) * 128 + x0 + p;
    out[o1] = sE[c * 64 + p] * fw + vf[o1];
    out[o2] = sE[(64 + c) * 64 + p] * fw + vf[o2];
  }
}

// ---------------------------------------------------------------------------
extern "C" void kernel_launch(void* const* d_in, const int* in_sizes, int n_in,
                              void* d_out, int out_size, void* d_ws, size_t ws_size,
                              hipStream_t stream) {
  const float* vf    = (const float*)d_in[0];
  const float* mw1   = (const float*)d_in[1];
  const float* mb1   = (const float*)d_in[2];
  const float* mw2   = (const float*)d_in[3];
  const float* mb2   = (const float*)d_in[4];
  const float* aw1   = (const float*)d_in[5];
  const float* ab1   = (const float*)d_in[6];
  const float* aw2   = (const float*)d_in[7];
  const float* ab2   = (const float*)d_in[8];
  const float* wqkv  = (const float*)d_in[9];
  const float* bqkv  = (const float*)d_in[10];
  const float* wo    = (const float*)d_in[11];
  const float* bo    = (const float*)d_in[12];
  const float* fcw   = (const float*)d_in[13];
  const float* fcb   = (const float*)d_in[14];
  float* out = (float*)d_out;

  // Workspace layout (~101.1 MiB):
  //   [0, 1MiB)    packed f16 weights: w1p | w2p | wqkvp | wop
  //   [1, 9MiB)    t1   : (B,2,32,128,128) f16
  //   [9, 41MiB)   comb : (B,128,128,128) f32  (blended in place -> enhanced)
  //   [41, ~101MiB) att : (961,4,128,8,8) f16
  char* ws = (char*)d_ws;
  _Float16* w1p   = (_Float16*)(ws + 0);
  _Float16* w2p   = (_Float16*)(ws + 73728);
  _Float16* wqkvp = (_Float16*)(ws + 147456);
  _Float16* wop   = (_Float16*)(ws + 245760);
  _Float16* t1    = (_Float16*)(ws + ((size_t)1 << 20));
  float*    comb  = (float*)(ws + ((size_t)9 << 20));
  _Float16* att   = (_Float16*)(ws + ((size_t)41 << 20));

  prepack_kernel<<<dim3((139264 + 255) / 256), 256, 0, stream>>>(
      mw1, aw1, mw2, aw2, wqkv, wo, w1p, w2p, wqkvp, wop);
  conv1_kernel<<<dim3(256, 2, 4), 256, 0, stream>>>(vf, w1p, mb1, ab1, t1);
  conv2_kernel<<<dim3(256, 2, 4), 256, 0, stream>>>(t1, w2p, mb2, ab2, comb);
  attn_kernel<<<dim3(961 * 4), 256, 0, stream>>>(comb, wqkvp, bqkv, wop, bo, att);
  blend_kernel<<<dim3((4u * 128 * 128 * 128) / 256), 256, 0, stream>>>(comb, att);
  final_kernel<<<dim3(256, 4), 256, 0, stream>>>(comb, vf, fcw, fcb, out);
}
